// SelfAttention_25649544691855
// MI455X (gfx1250) — matrix operand order repaired and checked by reading, not compiled, not run on hardware
//
#include <hip/hip_runtime.h>
#include <hip/hip_bf16.h>
#include <math.h>

typedef __attribute__((ext_vector_type(16))) __bf16 v16bf;
typedef __attribute__((ext_vector_type(8)))  float  v8f;
typedef __attribute__((ext_vector_type(4)))  unsigned int tdm_v4u;
typedef __attribute__((ext_vector_type(8)))  int          tdm_v8i;
typedef __attribute__((ext_vector_type(4)))  int          tdm_v4i;

#define NHEAD   32
#define SEQ     2048
#define EMB     2048
#define HD      64
#define MROWS   4096      // B*S
#define NQKV    6144

// ---------------------------------------------------------------------------
// WMMA fragment loads from LDS (CDNA5 16-bit layouts, wave32)
// A 16x32: lanes 0-15 -> K{0..15}, lanes 16-31 -> K{16..31} (contiguous, same order as B)
// B 32x16: lane -> N=lane&15, 16 contiguous K at (lane>>4)*16
// ---------------------------------------------------------------------------
static __device__ __forceinline__ v16bf load_a16(const __bf16* p0, const __bf16* p1) {
  v16bf a;
#pragma unroll
  for (int i = 0; i < 8; ++i) { a[i] = p0[i]; a[i + 8] = p1[i]; }
  return a;
}
static __device__ __forceinline__ v16bf load_b16(const __bf16* p) {
  v16bf b;
#pragma unroll
  for (int i = 0; i < 16; ++i) b[i] = p[i];
  return b;
}

// ---------------------------------------------------------------------------
// TDM: DMA a 2D tile (2-byte elements) global -> LDS with row padding.
// Descriptor packing per CDNA5 ISA.
//   w, h     : tile width (elems per row) / rows; tensor dims set == tile dims
//   stride   : row stride in the global tensor (elements)
//   pint     : pad_interval code (pad after 2^(pint+1) DWORDs)
//   pamt     : pad_amount code  (insert pamt+1 DWORDs)
// Toolchain here uses the 6-arg builtin form:
//   (uint32x4 g0, int32x8 g1, int32x4, int32x4, int32x8, i32 cpol)
// ---------------------------------------------------------------------------
static __device__ __forceinline__ void tdm_load_2d(
    unsigned lds_off, const void* gaddr,
    unsigned w, unsigned h, unsigned stride,
    unsigned pint, unsigned pamt)
{
  const unsigned long long ga = (unsigned long long)gaddr;
  tdm_v4u g0;
  g0[0] = 1u;                                             // count=1, user mode
  g0[1] = lds_off;                                        // lds_addr
  g0[2] = (unsigned)ga;                                   // global_addr lo
  g0[3] = (unsigned)((ga >> 32) & 0x01ffffffu) | 0x80000000u;  // addr hi | type=2
  tdm_v8i g1;
  g1[0] = (int)((1u << 16)            // data_size = 1 (2 bytes)
              | (1u << 20)            // pad_enable
              | (pint << 22) | (pamt << 25));
  g1[1] = (int)((w & 0xffffu) << 16);                     // tensor_dim0 lo -> bits 63:48
  g1[2] = (int)((w >> 16) | ((h & 0xffffu) << 16));       // dim0 hi | tensor_dim1 lo
  g1[3] = (int)((h >> 16) | ((w & 0xffffu) << 16));       // dim1 hi | tile_dim0
  g1[4] = (int)(h & 0xffffu);                             // tile_dim1 (tile_dim2=0)
  g1[5] = (int)stride;                                    // tensor_dim0_stride lo
  g1[6] = 0;
  g1[7] = 0;
  tdm_v4i z4 = {0, 0, 0, 0};
  tdm_v8i z8 = {0, 0, 0, 0, 0, 0, 0, 0};
  __builtin_amdgcn_tensor_load_to_lds(g0, g1, z4, z4, z8, 0);
}

// ---------------------------------------------------------------------------
// fp32 -> bf16 elementwise conversion (for x)
// ---------------------------------------------------------------------------
__global__ __launch_bounds__(256) void cvt_bf16_kernel(const float* __restrict__ src,
                                                       __bf16* __restrict__ dst, int n) {
  int i = blockIdx.x * blockDim.x + threadIdx.x;
  const int stride = gridDim.x * blockDim.x;
  for (; i < n; i += stride) dst[i] = (__bf16)src[i];
}

// ---------------------------------------------------------------------------
// fp32 [rows][cols] -> bf16 [cols][rows]  (weight pre-transpose, 32x32 tiles)
// ---------------------------------------------------------------------------
__global__ __launch_bounds__(256) void transpose_cvt_kernel(
    const float* __restrict__ src, __bf16* __restrict__ dst, int rows, int cols)
{
  __shared__ float t[32][33];
  const int r0 = blockIdx.y * 32;
  const int c0 = blockIdx.x * 32;
  const int tx = threadIdx.x & 31;
  const int ty = threadIdx.x >> 5;   // 0..7
#pragma unroll
  for (int i = 0; i < 32; i += 8)
    t[ty + i][tx] = src[(size_t)(r0 + ty + i) * cols + c0 + tx];
  __syncthreads();
#pragma unroll
  for (int i = 0; i < 32; i += 8)
    dst[(size_t)(c0 + ty + i) * rows + r0 + tx] = (__bf16)t[tx][ty + i];
}

// ---------------------------------------------------------------------------
// GEMM1: qkv = x @ Wqkv + bqkv, fused partial-RoPE epilogue.
// A [4096][2048] bf16, Bt [6144][2048] bf16 (pre-transposed).
// TDM double-buffered tiles; 8 waves, each 32x64 via 2x4 wmma accumulators.
// Outputs: q,k as [B,H,S,HD] bf16 ; v as [B,H,HD,S] bf16 (pre-transposed).
// ---------------------------------------------------------------------------
__global__ __launch_bounds__(256) void gemm_qkv_kernel(
    const __bf16* __restrict__ A,
    const __bf16* __restrict__ Bt,
    const float*  __restrict__ bias,
    __bf16* __restrict__ qbuf, __bf16* __restrict__ kbuf, __bf16* __restrict__ vtbuf)
{
  __shared__ __bf16 As[2][128][40];   // rows padded 32 -> 40 via TDM pad
  __shared__ __bf16 Bs[2][128][40];
  const int tile_n = blockIdx.x * 128;
  const int tile_m = blockIdx.y * 128;
  const int tid  = threadIdx.x;
  const int lane = tid & 31;
  const int wave = tid >> 5;
  const int wm = wave & 3;
  const int wn = wave >> 2;
  const int nlane = lane & 15;
  const int hq = lane >> 4;

  v8f acc[2][4];
#pragma unroll
  for (int mt = 0; mt < 2; ++mt)
#pragma unroll
    for (int nt = 0; nt < 4; ++nt)
#pragma unroll
      for (int i = 0; i < 8; ++i) acc[mt][nt][i] = 0.0f;

  if (wave == 0) {
    tdm_load_2d((unsigned)(size_t)&As[0][0][0], A  + (size_t)tile_m * EMB, 32, 128, EMB, 3, 3);
    tdm_load_2d((unsigned)(size_t)&Bs[0][0][0], Bt + (size_t)tile_n * EMB, 32, 128, EMB, 3, 3);
  }

  for (int kt = 0; kt < EMB / 32; ++kt) {
    const int cur = kt & 1;
    if (wave == 0) {
      if (kt + 1 < EMB / 32) {
        const int nxt = cur ^ 1;
        const int k1 = (kt + 1) * 32;
        tdm_load_2d((unsigned)(size_t)&As[nxt][0][0],
                    A  + (size_t)tile_m * EMB + k1, 32, 128, EMB, 3, 3);
        tdm_load_2d((unsigned)(size_t)&Bs[nxt][0][0],
                    Bt + (size_t)tile_n * EMB + k1, 32, 128, EMB, 3, 3);
        __builtin_amdgcn_s_wait_tensorcnt(2);
      } else {
        __builtin_amdgcn_s_wait_tensorcnt(0);
      }
    }
    __syncthreads();

    v16bf bfrag[4];
#pragma unroll
    for (int nt = 0; nt < 4; ++nt)
      bfrag[nt] = load_b16(&Bs[cur][wn * 64 + nt * 16 + nlane][hq * 16]);
#pragma unroll
    for (int mt = 0; mt < 2; ++mt) {
      const __bf16* p0 = &As[cur][wm * 32 + mt * 16 + nlane][hq * 16];
      v16bf a = load_a16(p0, p0 + 8);
#pragma unroll
      for (int nt = 0; nt < 4; ++nt)
        acc[mt][nt] = __builtin_amdgcn_wmma_f32_16x16x32_bf16(
            false, a, false, bfrag[nt], (short)0, acc[mt][nt], false, false);
    }
    __syncthreads();
  }

  // epilogue: bias + RoPE (first 32 dims of each head) + scatter
  const int colblock = tile_n + wn * 64;    // one (which, head) chunk
  const int which = colblock >> 11;         // 0=q 1=k 2=v
  const int h = (colblock >> 6) & 31;
  const float invf = __expf(-((2.0f * (float)nlane) / 32.0f) * 9.2103403719761836f);
  float biasv[4];
#pragma unroll
  for (int nt = 0; nt < 4; ++nt) biasv[nt] = bias[colblock + nt * 16 + nlane];

#pragma unroll
  for (int mt = 0; mt < 2; ++mt) {
#pragma unroll
    for (int r = 0; r < 8; ++r) {
      const int gm = tile_m + wm * 32 + mt * 16 + hq * 8 + r;   // token row
      const int bidx = gm >> 11;
      const int s = gm & 2047;
      float v0 = acc[mt][0][r] + biasv[0];
      float v1 = acc[mt][1][r] + biasv[1];
      float v2 = acc[mt][2][r] + biasv[2];
      float v3 = acc[mt][3][r] + biasv[3];
      if (which < 2) {
        float sn, cs;
        __sincosf((float)s * invf, &sn, &cs);
        const float t1 = v0, t2 = v1;
        v0 = t1 * cs - t2 * sn;
        v1 = t1 * sn + t2 * cs;
        __bf16* dst = (which == 0) ? qbuf : kbuf;
        const size_t base = (((size_t)bidx * NHEAD + h) * SEQ + s) * HD;
        dst[base + 0  + nlane] = (__bf16)v0;
        dst[base + 16 + nlane] = (__bf16)v1;
        dst[base + 32 + nlane] = (__bf16)v2;
        dst[base + 48 + nlane] = (__bf16)v3;
      } else {
        const size_t vb = ((size_t)bidx * NHEAD + h) * HD;
        vtbuf[(vb + 0  + nlane) * SEQ + s] = (__bf16)v0;
        vtbuf[(vb + 16 + nlane) * SEQ + s] = (__bf16)v1;
        vtbuf[(vb + 32 + nlane) * SEQ + s] = (__bf16)v2;
        vtbuf[(vb + 48 + nlane) * SEQ + s] = (__bf16)v3;
      }
    }
  }
}

// ---------------------------------------------------------------------------
// Flash attention: one (batch,head,64-query block) per 128-thread block.
// K/V tiles TDM double-buffered; each of 4 waves owns a 16-row query strip.
// ---------------------------------------------------------------------------
__global__ __launch_bounds__(128) void attn_kernel(
    const __bf16* __restrict__ qbuf, const __bf16* __restrict__ kbuf,
    const __bf16* __restrict__ vtbuf, __bf16* __restrict__ obuf)
{
  __shared__ __bf16 Ks[2][64][72];    // [key][d], rows padded 64 -> 72 via TDM
  __shared__ __bf16 Vs[2][64][72];    // [d][key] (global V pre-transposed)
  __shared__ float  Sc[4][16][65];    // per-wave score strips
  __shared__ __bf16 Pb[4][16][72];    // per-wave P strips (bf16)
  __shared__ float  mrow[64], lrow[64], crow[64];

  const int bh = blockIdx.x >> 5;     // b*H + h
  const int qblk = blockIdx.x & 31;
  const int bg = bh >> 5;
  const int hh = bh & 31;
  const int tid = threadIdx.x;
  const int lane = tid & 31;
  const int wave = tid >> 5;          // 0..3
  const int nlane = lane & 15;
  const int hq = lane >> 4;
  const float scale = 0.125f;         // hd^-0.5

  const size_t qkbase = (size_t)bh * SEQ * HD;
  const size_t vbase  = (size_t)bh * HD * SEQ;

  if (tid < 64) { mrow[tid] = -1e30f; lrow[tid] = 0.0f; }

  // Q fragments straight from global (used once per block)
  v16bf aQ[2];
  {
    const __bf16* qrow = qbuf + qkbase + (size_t)(qblk * 64 + wave * 16 + nlane) * HD;
#pragma unroll
    for (int kk = 0; kk < 2; ++kk)
      aQ[kk] = load_a16(qrow + kk * 32 + hq * 16, qrow + kk * 32 + 8 + hq * 16);
  }

  v8f Oa[4];
#pragma unroll
  for (int nt = 0; nt < 4; ++nt)
#pragma unroll
    for (int i = 0; i < 8; ++i) Oa[nt][i] = 0.0f;

  if (wave == 0) {
    tdm_load_2d((unsigned)(size_t)&Ks[0][0][0], kbuf + qkbase, 64, 64, 64, 4, 3);
    tdm_load_2d((unsigned)(size_t)&Vs[0][0][0], vtbuf + vbase, 64, 64, SEQ, 4, 3);
  }

  for (int j = 0; j <= qblk; ++j) {
    const int cur = j & 1;
    if (wave == 0) {
      if (j < qblk) {
        const int nxt = cur ^ 1;
        tdm_load_2d((unsigned)(size_t)&Ks[nxt][0][0],
                    kbuf + qkbase + (size_t)(j + 1) * 64 * HD, 64, 64, 64, 4, 3);
        tdm_load_2d((unsigned)(size_t)&Vs[nxt][0][0],
                    vtbuf + vbase + (size_t)(j + 1) * 64, 64, 64, SEQ, 4, 3);
        __builtin_amdgcn_s_wait_tensorcnt(2);
      } else {
        __builtin_amdgcn_s_wait_tensorcnt(0);
      }
    }
    __syncthreads();

    // S strip = Q_strip x K^T (16x64), scaled
#pragma unroll
    for (int nt = 0; nt < 4; ++nt) {
      v8f sc;
#pragma unroll
      for (int i = 0; i < 8; ++i) sc[i] = 0.0f;
#pragma unroll
      for (int kk = 0; kk < 2; ++kk) {
        v16bf bK = load_b16(&Ks[cur][nt * 16 + nlane][kk * 32 + hq * 16]);
        sc = __builtin_amdgcn_wmma_f32_16x16x32_bf16(
            false, aQ[kk], false, bK, (short)0, sc, false, false);
      }
#pragma unroll
      for (int r = 0; r < 8; ++r)
        Sc[wave][hq * 8 + r][nt * 16 + nlane] = sc[r] * scale;
    }
    __syncthreads();

    // online softmax: lanes 0-15 each own one query row of this wave's strip
    if (lane < 16) {
      const int row = lane;
      const int qglob = qblk * 64 + wave * 16 + row;
      const int ri = wave * 16 + row;
      const float m_old = mrow[ri];
      float mnew = m_old;
      for (int i = 0; i < 64; ++i) {
        if (j * 64 + i <= qglob) mnew = fmaxf(mnew, Sc[wave][row][i]);
      }
      const float corr = __expf(m_old - mnew);
      float lsum = 0.0f;
      for (int i = 0; i < 64; ++i) {
        float p = 0.0f;
        if (j * 64 + i <= qglob) p = __expf(Sc[wave][row][i] - mnew);
        lsum += p;
        Pb[wave][row][i] = (__bf16)p;
      }
      mrow[ri] = mnew;
      lrow[ri] = lrow[ri] * corr + lsum;
      crow[ri] = corr;
    }
    __syncthreads();

    // rescale running O, then O += P x V
    float cf[8];
#pragma unroll
    for (int r = 0; r < 8; ++r) cf[r] = crow[wave * 16 + hq * 8 + r];
#pragma unroll
    for (int nt = 0; nt < 4; ++nt)
#pragma unroll
      for (int r = 0; r < 8; ++r) Oa[nt][r] *= cf[r];

    v16bf aP[2];
#pragma unroll
    for (int kk = 0; kk < 2; ++kk) {
      const __bf16* p0 = &Pb[wave][nlane][kk * 32 + hq * 16];
      aP[kk] = load_a16(p0, p0 + 8);
    }
#pragma unroll
    for (int nt = 0; nt < 4; ++nt) {
#pragma unroll
      for (int kk = 0; kk < 2; ++kk) {
        v16bf bV = load_b16(&Vs[cur][nt * 16 + nlane][kk * 32 + hq * 16]);
        Oa[nt] = __builtin_amdgcn_wmma_f32_16x16x32_bf16(
            false, aP[kk], false, bV, (short)0, Oa[nt], false, false);
      }
    }
    __syncthreads();
  }

  // final normalize + store as [B,S,H,HD] bf16 (== [B,S,E] for GEMM2)
  float linv[8];
#pragma unroll
  for (int r = 0; r < 8; ++r) linv[r] = 1.0f / lrow[wave * 16 + hq * 8 + r];
#pragma unroll
  for (int nt = 0; nt < 4; ++nt) {
    const int d = nt * 16 + nlane;
#pragma unroll
    for (int r = 0; r < 8; ++r) {
      const int srow = qblk * 64 + wave * 16 + hq * 8 + r;
      obuf[(((size_t)bg * SEQ + srow) * NHEAD + hh) * HD + d] =
          (__bf16)(Oa[nt][r] * linv[r]);
    }
  }
}

// ---------------------------------------------------------------------------
// GEMM2: out = o @ Wo + bo  (fp32 output). Bt = Wo pre-transposed [N][K].
// ---------------------------------------------------------------------------
__global__ __launch_bounds__(256) void gemm_out_kernel(
    const __bf16* __restrict__ A,
    const __bf16* __restrict__ Bt,
    const float*  __restrict__ bias,
    float* __restrict__ out)
{
  __shared__ __bf16 As[2][128][40];
  __shared__ __bf16 Bs[2][128][40];
  const int tile_n = blockIdx.x * 128;
  const int tile_m = blockIdx.y * 128;
  const int tid  = threadIdx.x;
  const int lane = tid & 31;
  const int wave = tid >> 5;
  const int wm = wave & 3;
  const int wn = wave >> 2;
  const int nlane = lane & 15;
  const int hq = lane >> 4;

  v8f acc[2][4];
#pragma unroll
  for (int mt = 0; mt < 2; ++mt)
#pragma unroll
    for (int nt = 0; nt < 4; ++nt)
#pragma unroll
      for (int i = 0; i < 8; ++i) acc[mt][nt][i] = 0.0f;

  if (wave == 0) {
    tdm_load_2d((unsigned)(size_t)&As[0][0][0], A  + (size_t)tile_m * EMB, 32, 128, EMB, 3, 3);
    tdm_load_2d((unsigned)(size_t)&Bs[0][0][0], Bt + (size_t)tile_n * EMB, 32, 128, EMB, 3, 3);
  }

  for (int kt = 0; kt < EMB / 32; ++kt) {
    const int cur = kt & 1;
    if (wave == 0) {
      if (kt + 1 < EMB / 32) {
        const int nxt = cur ^ 1;
        const int k1 = (kt + 1) * 32;
        tdm_load_2d((unsigned)(size_t)&As[nxt][0][0],
                    A  + (size_t)tile_m * EMB + k1, 32, 128, EMB, 3, 3);
        tdm_load_2d((unsigned)(size_t)&Bs[nxt][0][0],
                    Bt + (size_t)tile_n * EMB + k1, 32, 128, EMB, 3, 3);
        __builtin_amdgcn_s_wait_tensorcnt(2);
      } else {
        __builtin_amdgcn_s_wait_tensorcnt(0);
      }
    }
    __syncthreads();

    v16bf bfrag[4];
#pragma unroll
    for (int nt = 0; nt < 4; ++nt)
      bfrag[nt] = load_b16(&Bs[cur][wn * 64 + nt * 16 + nlane][hq * 16]);
#pragma unroll
    for (int mt = 0; mt < 2; ++mt) {
      const __bf16* p0 = &As[cur][wm * 32 + mt * 16 + nlane][hq * 16];
      v16bf a = load_a16(p0, p0 + 8);
#pragma unroll
      for (int nt = 0; nt < 4; ++nt)
        acc[mt][nt] = __builtin_amdgcn_wmma_f32_16x16x32_bf16(
            false, a, false, bfrag[nt], (short)0, acc[mt][nt], false, false);
    }
    __syncthreads();
  }

#pragma unroll
  for (int mt = 0; mt < 2; ++mt)
#pragma unroll
    for (int nt = 0; nt < 4; ++nt) {
      const int col = tile_n + wn * 64 + nt * 16 + nlane;
      const float bv = bias[col];
#pragma unroll
      for (int r = 0; r < 8; ++r) {
        const int gm = tile_m + wm * 32 + mt * 16 + hq * 8 + r;
        out[(size_t)gm * EMB + col] = acc[mt][nt][r] + bv;
      }
    }
}

// ---------------------------------------------------------------------------
extern "C" void kernel_launch(void* const* d_in, const int* in_sizes, int n_in,
                              void* d_out, int out_size, void* d_ws, size_t ws_size,
                              hipStream_t stream) {
  const float* x    = (const float*)d_in[0];
  const float* Wqkv = (const float*)d_in[1];
  const float* bqkv = (const float*)d_in[2];
  const float* Wo   = (const float*)d_in[3];
  const float* bo   = (const float*)d_in[4];
  float* out = (float*)d_out;

  char* ws = (char*)d_ws;
  size_t off = 0;
  auto alloc = [&](size_t bytes) -> void* {
    void* p = ws + off;
    off += (bytes + 255) & ~(size_t)255;
    return p;
  };
  __bf16* xb    = (__bf16*)alloc((size_t)MROWS * EMB * 2);   // x bf16 [M][K]
  __bf16* wqkvt = (__bf16*)alloc((size_t)EMB * NQKV * 2);    // Wqkv^T bf16 [6144][2048]
  __bf16* wot   = (__bf16*)alloc((size_t)EMB * EMB * 2);     // Wo^T bf16 [2048][2048]
  __bf16* qb    = (__bf16*)alloc((size_t)MROWS * EMB * 2);   // [B,H,S,HD]
  __bf16* kb    = (__bf16*)alloc((size_t)MROWS * EMB * 2);   // [B,H,S,HD]
  __bf16* vtb   = (__bf16*)alloc((size_t)MROWS * EMB * 2);   // [B,H,HD,S]
  __bf16* ob    = (__bf16*)alloc((size_t)MROWS * EMB * 2);   // [B,S,E]

  cvt_bf16_kernel<<<2048, 256, 0, stream>>>(x, xb, MROWS * EMB);
  transpose_cvt_kernel<<<dim3(NQKV / 32, EMB / 32), 256, 0, stream>>>(
      Wqkv, wqkvt, EMB, NQKV);
  transpose_cvt_kernel<<<dim3(EMB / 32, EMB / 32), 256, 0, stream>>>(
      Wo, wot, EMB, EMB);

  gemm_qkv_kernel<<<dim3(NQKV / 128, MROWS / 128), 256, 0, stream>>>(
      xb, wqkvt, bqkv, qb, kb, vtb);

  attn_kernel<<<2 * NHEAD * (SEQ / 64), 128, 0, stream>>>(qb, kb, vtb, ob);

  gemm_out_kernel<<<dim3(EMB / 128, MROWS / 128), 256, 0, stream>>>(
      ob, wot, bo, out);
}
